// UNet_5695126634524
// MI455X (gfx1250) — matrix operand order repaired and checked by reading, not compiled, not run on hardware
//
#include <hip/hip_runtime.h>
#include <hip/hip_bf16.h>
#include <math.h>

typedef __attribute__((ext_vector_type(16))) _Float16     v16h;
typedef __attribute__((ext_vector_type(8)))  float        v8f;
typedef __attribute__((ext_vector_type(8)))  _Float16     v8h;
typedef __attribute__((ext_vector_type(4)))  unsigned int v4u;
typedef __attribute__((ext_vector_type(8)))  unsigned int v8u;
typedef _Float16 half_t;

#define BN_EPS 1e-5f

static __device__ __forceinline__ v16h frag_from(const half_t* p, unsigned mask) {
  v4u lo = *(const v4u*)p;
  v4u hi = *(const v4u*)(p + 8);
  lo &= mask; hi &= mask;
  v8u w;
#pragma unroll
  for (int j = 0; j < 4; ++j) { w[j] = lo[j]; w[4 + j] = hi[j]; }
  return __builtin_bit_cast(v16h, w);
}
static __device__ __forceinline__ v16h frag_from(const half_t* p) {
  v4u lo = *(const v4u*)p;
  v4u hi = *(const v4u*)(p + 8);
  v8u w;
#pragma unroll
  for (int j = 0; j < 4; ++j) { w[j] = lo[j]; w[4 + j] = hi[j]; }
  return __builtin_bit_cast(v16h, w);
}

// ---------------- prep kernels ----------------

// x: f32 NCHW [3,64,64] -> f16 NHWC [64*64, 32] zero-padded channels
__global__ void k_cvt_x(const float* __restrict__ in, half_t* __restrict__ out) {
  int i = blockIdx.x * blockDim.x + threadIdx.x;   // over 4096*32
  if (i >= 4096 * 32) return;
  int c = i & 31, sp = i >> 5;
  out[i] = (c < 3) ? (half_t)in[c * 4096 + sp] : (half_t)0.f;
}

// conv3x3 weights f32 [Cout,Cin,3,3] -> f16 [CW, 9*Cp] tap-major (k = tap*Cp + ci), zero-padded
__global__ void k_cvt_w3(const float* __restrict__ in, half_t* __restrict__ out,
                         int Cout, int Cin, int Cp, int CW) {
  int Kp = 9 * Cp;
  int n = CW * Kp;
  int i = blockIdx.x * blockDim.x + threadIdx.x;
  int stride = gridDim.x * blockDim.x;
  for (; i < n; i += stride) {
    int r = i / Kp, rem = i - r * Kp;
    int tap = rem / Cp, ci = rem - tap * Cp;
    out[i] = (r < Cout && ci < Cin) ? (half_t)in[((size_t)r * Cin + ci) * 9 + tap]
                                    : (half_t)0.f;
  }
}

// 1x1 weights f32 [Cout,K] -> f16 [CW,Kp] zero-padded
__global__ void k_cvt_w1(const float* __restrict__ in, half_t* __restrict__ out,
                         int Cout, int K, int Kp, int CW) {
  int n = CW * Kp;
  int i = blockIdx.x * blockDim.x + threadIdx.x;
  int stride = gridDim.x * blockDim.x;
  for (; i < n; i += stride) {
    int r = i / Kp, k = i - r * Kp;
    out[i] = (r < Cout && k < K) ? (half_t)in[(size_t)r * K + k] : (half_t)0.f;
  }
}

// fold conv-bias + BN(eval): y = relu(alpha*conv + beta)
__global__ void k_fold_bn(const float* __restrict__ bias, const float* __restrict__ g,
                          const float* __restrict__ be,  const float* __restrict__ m,
                          const float* __restrict__ var, float* __restrict__ alpha,
                          float* __restrict__ beta, int C) {
  int c = blockIdx.x * blockDim.x + threadIdx.x;
  if (c < C) {
    float a = g[c] * rsqrtf(var[c] + BN_EPS);
    alpha[c] = a;
    beta[c]  = be[c] + (bias[c] - m[c]) * a;
  }
}

// ---------------- NHWC elementwise ----------------

__global__ void k_maxpool2(const half_t* __restrict__ in, half_t* __restrict__ out,
                           int CS, int coff, int Cp, int H, int W) {
  int Ho = H >> 1, Wo = W >> 1;
  int n = Cp * Ho * Wo;
  int i = blockIdx.x * blockDim.x + threadIdx.x;
  if (i >= n) return;
  int c = i % Cp; int t = i / Cp; int x = t % Wo; int y = t / Wo;
  const half_t* p = in + (size_t)(2 * y * W + 2 * x) * CS + coff + c;
  float v0 = (float)p[0], v1 = (float)p[CS];
  float v2 = (float)p[(size_t)W * CS], v3 = (float)p[(size_t)(W + 1) * CS];
  out[i] = (half_t)fmaxf(fmaxf(v0, v1), fmaxf(v2, v3));
}

__global__ void k_upsample2(const half_t* __restrict__ in, half_t* __restrict__ out,
                            int Cin, int CS, int n) {
  int N2 = 2 * n;
  int total = Cin * N2 * N2;
  int i = blockIdx.x * blockDim.x + threadIdx.x;
  if (i >= total) return;
  int c = i % Cin; int pix = i / Cin; int jx = pix % N2; int jy = pix / N2;
  float scale = (float)(n - 1) / (float)(2 * n - 1);
  float cy = jy * scale, cx = jx * scale;
  int y0 = (int)cy, x0 = (int)cx;
  int y1 = min(y0 + 1, n - 1), x1 = min(x0 + 1, n - 1);
  float wy = cy - (float)y0, wx = cx - (float)x0;
  float v00 = (float)in[(size_t)(y0 * n + x0) * Cin + c];
  float v01 = (float)in[(size_t)(y0 * n + x1) * Cin + c];
  float v10 = (float)in[(size_t)(y1 * n + x0) * Cin + c];
  float v11 = (float)in[(size_t)(y1 * n + x1) * Cin + c];
  float vt = v00 * (1.f - wx) + v01 * wx;
  float vb = v10 * (1.f - wx) + v11 * wx;
  out[(size_t)pix * CS + c] = (half_t)(vt * (1.f - wy) + vb * wy);
}

// ---------------- WMMA implicit-GEMM conv3x3 + folded BN + ReLU (NHWC) ----------------
// 2x2 register-blocked: one wave owns {oc0, oc0+16} x {sp0, sp0+16}; each k-chunk does
// 8 b128 loads feeding 4 WMMAs (A and B fragments each reused twice).
__global__ void __launch_bounds__(32) k_conv3x3_wmma(
    const half_t* __restrict__ act, const half_t* __restrict__ wgt,
    const float* __restrict__ alpha, const float* __restrict__ beta,
    half_t* __restrict__ out, int Cp, int Cout, int CS, int coff,
    int H, int W, int wlog) {
  const int lane = threadIdx.x;
  const int ncol = lane & 15;
  const int hi   = lane >> 4;
  const int sp0  = blockIdx.x << 5;
  const int oc0  = blockIdx.y << 5;
  const int Kp   = 9 * Cp;
  const int spA  = sp0 + ncol,  spB = sp0 + 16 + ncol;
  const int oyA  = spA >> wlog, oxA = spA & (W - 1);
  const int oyB  = spB >> wlog, oxB = spB & (W - 1);
  const half_t* w0 = wgt + (size_t)(oc0 + ncol) * Kp + (hi << 4);
  const half_t* w1 = w0 + (size_t)16 * Kp;

  v8f acc00 = {}, acc01 = {}, acc10 = {}, acc11 = {};
  int kw = 0;
  for (int tap = 0; tap < 9; ++tap) {
    int dy = (tap * 11) >> 5;           // tap / 3
    int dx = tap - dy * 3;
    int iyA = oyA + dy - 1, ixA = oxA + dx - 1;
    int iyB = oyB + dy - 1, ixB = oxB + dx - 1;
    unsigned mA = (((unsigned)iyA < (unsigned)H) && ((unsigned)ixA < (unsigned)W)) ? ~0u : 0u;
    unsigned mB = (((unsigned)iyB < (unsigned)H) && ((unsigned)ixB < (unsigned)W)) ? ~0u : 0u;
    const half_t* rowA = act + (size_t)(min(max(iyA, 0), H - 1) * W + min(max(ixA, 0), W - 1)) * Cp + (hi << 4);
    const half_t* rowB = act + (size_t)(min(max(iyB, 0), H - 1) * W + min(max(ixB, 0), W - 1)) * Cp + (hi << 4);
    for (int c0 = 0; c0 < Cp; c0 += 32, kw += 32) {
      v16h a0 = frag_from(w0 + kw);
      v16h a1 = frag_from(w1 + kw);
      v16h bA = frag_from(rowA + c0, mA);
      v16h bB = frag_from(rowB + c0, mB);
      acc00 = __builtin_amdgcn_wmma_f32_16x16x32_f16(false, a0, false, bA, (short)0, acc00, false, false);
      acc01 = __builtin_amdgcn_wmma_f32_16x16x32_f16(false, a0, false, bB, (short)0, acc01, false, false);
      acc10 = __builtin_amdgcn_wmma_f32_16x16x32_f16(false, a1, false, bA, (short)0, acc10, false, false);
      acc11 = __builtin_amdgcn_wmma_f32_16x16x32_f16(false, a1, false, bB, (short)0, acc11, false, false);
    }
  }
#pragma unroll
  for (int t = 0; t < 2; ++t) {
    int cb = oc0 + (t << 4) + (hi << 3);
#pragma unroll
    for (int s = 0; s < 2; ++s) {
      v8f acc = t ? (s ? acc11 : acc10) : (s ? acc01 : acc00);
      int spo = sp0 + (s << 4) + ncol;
      v8h o;
#pragma unroll
      for (int v = 0; v < 8; ++v) {
        int c = cb + v;
        int cc = min(c, Cout - 1);
        float y = fmaxf(acc[v] * alpha[cc] + beta[cc], 0.f);
        o[v] = (c < Cout) ? (half_t)y : (half_t)0.f;
      }
      *(v8h*)(out + (size_t)spo * CS + coff + cb) = o;
    }
  }
}

// ---------------- WMMA 1x1 conv + bias + sigmoid -> f32 NCHW (2x2 blocked) ----------------
__global__ void __launch_bounds__(32) k_conv1x1_sig_wmma(
    const half_t* __restrict__ act, const half_t* __restrict__ wgt,
    const float* __restrict__ bias, float* __restrict__ out,
    int Cp, int Cout, int HW) {
  const int lane = threadIdx.x;
  const int ncol = lane & 15;
  const int hi   = lane >> 4;
  const int sp0  = blockIdx.x << 5;
  const int oc0  = blockIdx.y << 5;
  const half_t* w0 = wgt + (size_t)(oc0 + ncol) * Cp + (hi << 4);
  const half_t* w1 = w0 + (size_t)16 * Cp;
  const half_t* rA = act + (size_t)(sp0 + ncol) * Cp + (hi << 4);
  const half_t* rB = rA + (size_t)16 * Cp;

  v8f acc00 = {}, acc01 = {}, acc10 = {}, acc11 = {};
  for (int k0 = 0; k0 < Cp; k0 += 32) {
    v16h a0 = frag_from(w0 + k0);
    v16h a1 = frag_from(w1 + k0);
    v16h bA = frag_from(rA + k0);
    v16h bB = frag_from(rB + k0);
    acc00 = __builtin_amdgcn_wmma_f32_16x16x32_f16(false, a0, false, bA, (short)0, acc00, false, false);
    acc01 = __builtin_amdgcn_wmma_f32_16x16x32_f16(false, a0, false, bB, (short)0, acc01, false, false);
    acc10 = __builtin_amdgcn_wmma_f32_16x16x32_f16(false, a1, false, bA, (short)0, acc10, false, false);
    acc11 = __builtin_amdgcn_wmma_f32_16x16x32_f16(false, a1, false, bB, (short)0, acc11, false, false);
  }
#pragma unroll
  for (int t = 0; t < 2; ++t) {
    int cb = oc0 + (t << 4) + (hi << 3);
#pragma unroll
    for (int s = 0; s < 2; ++s) {
      v8f acc = t ? (s ? acc11 : acc10) : (s ? acc01 : acc00);
      int spo = sp0 + (s << 4) + ncol;
#pragma unroll
      for (int v = 0; v < 8; ++v) {
        int c = cb + v;
        float y = acc[v] + bias[c];
        out[(size_t)c * HW + spo] = 1.f / (1.f + expf(-y));
      }
    }
  }
}

// ---------------- tail: shared 1->1 conv3x3+BN+ReLU per channel, sum, 1x1 sigmoid ----
__global__ void k_tail(const float* __restrict__ out1, const float* __restrict__ w9,
                       const float* __restrict__ alpha1, const float* __restrict__ beta1,
                       const float* __restrict__ w2, const float* __restrict__ b2,
                       float* __restrict__ out2) {
  int p = blockIdx.x * blockDim.x + threadIdx.x;
  if (p >= 4096) return;
  int iy = p >> 6, ix = p & 63;
  float a1 = alpha1[0], b1 = beta1[0];
  float wl[9];
#pragma unroll
  for (int t = 0; t < 9; ++t) wl[t] = w9[t];
  float s = 0.f;
  for (int c = 0; c < 256; ++c) {
    const float* ch = out1 + (size_t)c * 4096;
    float conv = 0.f;
#pragma unroll
    for (int dy = 0; dy < 3; ++dy) {
      int yy = iy + dy - 1;
      bool oky = (unsigned)yy < 64u;
      int yyc = min(max(yy, 0), 63);
#pragma unroll
      for (int dx = 0; dx < 3; ++dx) {
        int xx = ix + dx - 1;
        bool ok = oky && ((unsigned)xx < 64u);
        int xxc = min(max(xx, 0), 63);
        float v = ch[yyc * 64 + xxc];
        conv += wl[dy * 3 + dx] * (ok ? v : 0.f);
      }
    }
    s += fmaxf(a1 * conv + b1, 0.f);
  }
#pragma unroll
  for (int k = 0; k < 3; ++k)
    out2[k * 4096 + p] = 1.f / (1.f + expf(-(w2[k] * s + b2[k])));
}

// ---------------- host orchestration ----------------

extern "C" void kernel_launch(void* const* d_in, const int* in_sizes, int n_in,
                              void* d_out, int out_size, void* d_ws, size_t ws_size,
                              hipStream_t stream) {
  (void)n_in; (void)out_size; (void)ws_size;
  static const int CIN[15]  = {3,300,300,512,512,1024,1024,2048,3072,2048,1536,1024,812,512,1};
  static const int COUT[15] = {300,300,512,512,1024,1024,2048,2048,2048,1024,1024,512,512,300,1};
  static const int CPIN[15] = {32,320,320,512,512,1024,1024,2048,3072,2048,1536,1024,832,512,0};

  // ----- input index maps (insertion order vs jax sorted-key flatten) -----
  int wI[15], bI[15], gI[15], beI[15], mI[15], vI[15];
  int outW, outB, t2W, t2B, xI;
  if (in_sizes[0] == 3 * 64 * 64) {
    xI = 0;
    for (int b = 0; b < 15; ++b) {
      int s = 1 + 6 * b;
      wI[b] = s; bI[b] = s + 1; gI[b] = s + 2; beI[b] = s + 3; mI[b] = s + 4; vI[b] = s + 5;
    }
    outW = 91; outB = 92; t2W = 93; t2B = 94;
  } else {
    static const int START[15] = {0,6,12,18,24,30,36,42,82,88,70,76,58,64,50};
    for (int b = 0; b < 15; ++b) {
      int s = START[b];
      beI[b] = s; gI[b] = s + 1; mI[b] = s + 2; vI[b] = s + 3; bI[b] = s + 4; wI[b] = s + 5;
    }
    outB = 48; outW = 49; t2B = 56; t2W = 57; xI = 94;
  }

  int CW[15];
  for (int b = 0; b < 15; ++b) CW[b] = (COUT[b] + 31) & ~31;

  // ----- workspace carve-up -----
  char* wsp = (char*)d_ws;
  auto alloc = [&](size_t bytes) -> void* {
    void* r = (void*)wsp;
    wsp += (bytes + 255) & ~(size_t)255;
    return r;
  };

  half_t* wh[14]; float* al[15]; float* bt[15];
  for (int b = 0; b < 14; ++b)
    wh[b] = (half_t*)alloc((size_t)CW[b] * 9 * CPIN[b] * sizeof(half_t));
  for (int b = 0; b < 15; ++b) {
    al[b] = (float*)alloc((size_t)COUT[b] * sizeof(float));
    bt[b] = (float*)alloc((size_t)COUT[b] * sizeof(float));
  }
  half_t* owh  = (half_t*)alloc((size_t)256 * 320 * sizeof(half_t));
  half_t* ax   = (half_t*)alloc((size_t)4096 * 32 * 2);
  half_t* b1a  = (half_t*)alloc((size_t)4096 * 320 * 2);
  half_t* cat1 = (half_t*)alloc((size_t)4096 * 832 * 2);   // [up 0..511 | c1 512..811 | pad]
  half_t* p1   = (half_t*)alloc((size_t)1024 * 320 * 2);
  half_t* b2a  = (half_t*)alloc((size_t)1024 * 512 * 2);
  half_t* cat2 = (half_t*)alloc((size_t)1024 * 1536 * 2);  // [up 0..1023 | c2 1024..1535]
  half_t* p2   = (half_t*)alloc((size_t)256 * 512 * 2);
  half_t* b3a  = (half_t*)alloc((size_t)256 * 1024 * 2);
  half_t* cat3 = (half_t*)alloc((size_t)256 * 3072 * 2);   // [up 0..2047 | c3 2048..3071]
  half_t* p3   = (half_t*)alloc((size_t)64 * 1024 * 2);
  half_t* b4a  = (half_t*)alloc((size_t)64 * 2048 * 2);
  half_t* c4   = (half_t*)alloc((size_t)64 * 2048 * 2);
  half_t* u31o = (half_t*)alloc((size_t)256 * 2048 * 2);
  half_t* u32o = (half_t*)alloc((size_t)256 * 1024 * 2);
  half_t* u21o = (half_t*)alloc((size_t)1024 * 1024 * 2);
  half_t* u22o = (half_t*)alloc((size_t)1024 * 512 * 2);
  half_t* u11o = (half_t*)alloc((size_t)4096 * 512 * 2);
  half_t* u12o = (half_t*)alloc((size_t)4096 * 320 * 2);

  // ----- prep (re-done each launch: deterministic) -----
  k_cvt_x<<<(4096 * 32 + 255) / 256, 256, 0, stream>>>((const float*)d_in[xI], ax);
  for (int b = 0; b < 14; ++b) {
    int n = CW[b] * 9 * CPIN[b];
    int blocks = (n + 255) / 256; if (blocks > 8192) blocks = 8192;
    k_cvt_w3<<<blocks, 256, 0, stream>>>((const float*)d_in[wI[b]], wh[b],
                                         COUT[b], CIN[b], CPIN[b], CW[b]);
  }
  for (int b = 0; b < 15; ++b)
    k_fold_bn<<<(COUT[b] + 255) / 256, 256, 0, stream>>>(
        (const float*)d_in[bI[b]], (const float*)d_in[gI[b]], (const float*)d_in[beI[b]],
        (const float*)d_in[mI[b]], (const float*)d_in[vI[b]], al[b], bt[b], COUT[b]);
  k_cvt_w1<<<(256 * 320 + 255) / 256, 256, 0, stream>>>((const float*)d_in[outW], owh,
                                                        256, 300, 320, 256);

  auto conv3 = [&](const half_t* in, int blk, half_t* outp, int CS, int coff,
                   int H, int W, int wlog) {
    dim3 grid((H * W) / 32, CW[blk] / 32);
    k_conv3x3_wmma<<<grid, 32, 0, stream>>>(in, wh[blk], al[blk], bt[blk], outp,
                                            CPIN[blk], COUT[blk], CS, coff, H, W, wlog);
  };

  // ----- encoder -----
  conv3(ax,  0, b1a, 320, 0, 64, 64, 6);                               // d1a
  conv3(b1a, 1, cat1, 832, 512, 64, 64, 6);                            // d1b -> c1 region (+pad)
  k_maxpool2<<<(320 * 1024 + 255) / 256, 256, 0, stream>>>(cat1, p1, 832, 512, 320, 64, 64);
  conv3(p1,  2, b2a, 512, 0, 32, 32, 5);                               // d2a
  conv3(b2a, 3, cat2, 1536, 1024, 32, 32, 5);                          // d2b -> c2 region
  k_maxpool2<<<(512 * 256 + 255) / 256, 256, 0, stream>>>(cat2, p2, 1536, 1024, 512, 32, 32);
  conv3(p2,  4, b3a, 1024, 0, 16, 16, 4);                              // d3a
  conv3(b3a, 5, cat3, 3072, 2048, 16, 16, 4);                          // d3b -> c3 region
  k_maxpool2<<<(1024 * 64 + 255) / 256, 256, 0, stream>>>(cat3, p3, 3072, 2048, 1024, 16, 16);
  conv3(p3,  6, b4a, 2048, 0, 8, 8, 3);                                // d4a
  conv3(b4a, 7, c4, 2048, 0, 8, 8, 3);                                 // d4b

  // ----- decoder -----
  k_upsample2<<<(2048 * 256 + 255) / 256, 256, 0, stream>>>(c4, cat3, 2048, 3072, 8);
  conv3(cat3, 8, u31o, 2048, 0, 16, 16, 4);                            // u31
  conv3(u31o, 9, u32o, 1024, 0, 16, 16, 4);                            // u32
  k_upsample2<<<(1024 * 1024 + 255) / 256, 256, 0, stream>>>(u32o, cat2, 1024, 1536, 16);
  conv3(cat2, 10, u21o, 1024, 0, 32, 32, 5);                           // u21
  conv3(u21o, 11, u22o, 512, 0, 32, 32, 5);                            // u22
  k_upsample2<<<(512 * 4096 + 255) / 256, 256, 0, stream>>>(u22o, cat1, 512, 832, 32);
  conv3(cat1, 12, u11o, 512, 0, 64, 64, 6);                            // u11
  conv3(u11o, 13, u12o, 320, 0, 64, 64, 6);                            // u12

  // ----- heads -----
  float* out1 = (float*)d_out;
  dim3 g1(4096 / 32, 256 / 32);
  k_conv1x1_sig_wmma<<<g1, 32, 0, stream>>>(u12o, owh, (const float*)d_in[outB],
                                            out1, 320, 256, 4096);
  k_tail<<<16, 256, 0, stream>>>(out1, (const float*)d_in[wI[14]], al[14], bt[14],
                                 (const float*)d_in[t2W], (const float*)d_in[t2B],
                                 out1 + (size_t)256 * 4096);
}
